// MambaBlock_19464791785934
// MI455X (gfx1250) — matrix operand order repaired and checked by reading, not compiled, not run on hardware
//
#include <hip/hip_runtime.h>
#include <hip/hip_bf16.h>
#include <math.h>
#include <stdint.h>

typedef __bf16 bf16;
typedef __attribute__((ext_vector_type(16))) __bf16 bf16x16;
typedef __attribute__((ext_vector_type(8)))  __bf16 bf16x8;
typedef __attribute__((ext_vector_type(8)))  float  f32x8;
typedef __attribute__((ext_vector_type(4)))  int    i32x4;

static constexpr int kDIM     = 1024;
static constexpr int kDSTATE  = 128;
static constexpr int kDCONV   = 4;
static constexpr int kDINNER  = 2048;
static constexpr int kBATCH   = 4;
static constexpr int kSEQ     = 4096;
static constexpr int kTOKENS  = kBATCH * kSEQ;          // 16384
static constexpr int kNDBL    = 2 * kDSTATE + 1;        // 257
static constexpr int kLDA     = 40;                     // padded LDS row (32 + 8)

// --------------------------------------------------------------------------
// Async global->LDS copy (CDNA5 GLOBAL_LOAD_ASYNC_TO_LDS_B128, ASYNCcnt).
// Builtin signature (from clang diagnostic): (AS1 int4*, AS3 int4*, Ii, Ii).
// Guarded: falls back to VGPR-bounce staging if the builtins are absent.
// --------------------------------------------------------------------------
#if defined(__has_builtin)
#if __has_builtin(__builtin_amdgcn_global_load_async_to_lds_b128) && \
    __has_builtin(__builtin_amdgcn_s_wait_asynccnt)
#define USE_ASYNC_LDS 1
#endif
#endif
#ifndef USE_ASYNC_LDS
#define USE_ASYNC_LDS 0
#endif

__device__ __forceinline__ void cp16_g2l(const bf16* g, bf16* l) {
#if USE_ASYNC_LDS
    __builtin_amdgcn_global_load_async_to_lds_b128(
        (__attribute__((address_space(1))) i32x4*)(g),
        (__attribute__((address_space(3))) i32x4*)(l), 0, 0);
#else
    *(bf16x8*)l = *(const bf16x8*)g;
#endif
}

// ---------------------------------------------------------------------------
// f32 -> bf16 conversion
// ---------------------------------------------------------------------------
__global__ __launch_bounds__(256) void f32_to_bf16_kernel(const float* __restrict__ in,
                                                          bf16* __restrict__ out,
                                                          long long n) {
    long long i = (long long)blockIdx.x * 256 + threadIdx.x;
    if (i < n) out[i] = (bf16)in[i];
}

// ---------------------------------------------------------------------------
// WMMA bf16 GEMM:  C[M,N] = A[M,K] * W[N,K]^T   (A,W row-major bf16)
// Block: 256 threads = 8 waves (4 M x 2 N); block tile 128x64; K step 32.
// A/B tiles double-buffered in LDS, filled via async global->LDS copies.
// A frag (lane,e): M = lane%16,  K = (e<8?0:16) + 8*(lane/16) + (e&7)
// B frag (lane,e): N = lane%16,  K = (e<8?0:16) + 8*(lane/16) + (e&7)
// D frag (lane,v): M = v + 8*(lane/16), N = lane%16
// ---------------------------------------------------------------------------
template <bool OUT_BF16>
__global__ __launch_bounds__(256) void wmma_gemm_kernel(const bf16* __restrict__ A,
                                                        const bf16* __restrict__ W,
                                                        void* __restrict__ Cout,
                                                        int M, int N, int K) {
    __shared__ bf16 sA[2][128][kLDA];
    __shared__ bf16 sB[2][64][kLDA];

    const int lane = threadIdx.x & 31;
    const int wave = threadIdx.x >> 5;
    const int wm   = wave & 3;             // 0..3  (M)
    const int wn   = wave >> 2;            // 0..1  (N)
    const long long mblk = (long long)blockIdx.y * 128;
    const long long nblk = (long long)blockIdx.x * 64;
    const int half = lane >> 4;
    const int l16  = lane & 15;

    // global source coordinates for this thread's staging slices
    const int arow = threadIdx.x >> 1;               // 0..127
    const int acol = (threadIdx.x & 1) * 16;         // 0 or 16
    const int brow = threadIdx.x >> 2;               // 0..63
    const int bcol = (threadIdx.x & 3) * 8;          // 0,8,16,24
    const long long bsrc = nblk + brow;
    const bf16* gA = A + (mblk + arow) * (long long)K + acol;
    const bf16* gB = W + ((bsrc < N) ? bsrc : 0) * (long long)K + bcol;

    auto fill = [&](int buf, int k) {
        cp16_g2l(gA + k,     &sA[buf][arow][acol]);
        cp16_g2l(gA + k + 8, &sA[buf][arow][acol + 8]);
        cp16_g2l(gB + k,     &sB[buf][brow][bcol]);
        if (k + 64 < K) {                            // prefetch next-next tile
            __builtin_prefetch(gA + k + 64, 0, 0);
            __builtin_prefetch(gB + k + 64, 0, 0);
        }
    };

    f32x8 acc00 = {}, acc01 = {}, acc10 = {}, acc11 = {};
    const int KT = K >> 5;

    fill(0, 0);
    for (int kt = 0; kt < KT; ++kt) {
        const int cur = kt & 1;
        if (kt + 1 < KT) {
            fill(cur ^ 1, (kt + 1) << 5);
#if USE_ASYNC_LDS
            __builtin_amdgcn_s_wait_asynccnt(3);     // oldest fill (cur) done
#endif
        } else {
#if USE_ASYNC_LDS
            __builtin_amdgcn_s_wait_asynccnt(0);
#endif
        }
        __syncthreads();

        bf16x16 a0, a1, b0, b1;
        {
            const bf16* r = &sA[cur][wm * 32 + l16][8 * half];
            bf16x8 lo = *(const bf16x8*)r;
            bf16x8 hi = *(const bf16x8*)(r + 16);
#pragma unroll
            for (int e = 0; e < 8; ++e) { a0[e] = lo[e]; a0[8 + e] = hi[e]; }
        }
        {
            const bf16* r = &sA[cur][wm * 32 + 16 + l16][8 * half];
            bf16x8 lo = *(const bf16x8*)r;
            bf16x8 hi = *(const bf16x8*)(r + 16);
#pragma unroll
            for (int e = 0; e < 8; ++e) { a1[e] = lo[e]; a1[8 + e] = hi[e]; }
        }
        {
            const bf16* r = &sB[cur][wn * 32 + l16][8 * half];
            bf16x8 lo = *(const bf16x8*)r;
            bf16x8 hi = *(const bf16x8*)(r + 16);
#pragma unroll
            for (int e = 0; e < 8; ++e) { b0[e] = lo[e]; b0[8 + e] = hi[e]; }
        }
        {
            const bf16* r = &sB[cur][wn * 32 + 16 + l16][8 * half];
            bf16x8 lo = *(const bf16x8*)r;
            bf16x8 hi = *(const bf16x8*)(r + 16);
#pragma unroll
            for (int e = 0; e < 8; ++e) { b1[e] = lo[e]; b1[8 + e] = hi[e]; }
        }
        acc00 = __builtin_amdgcn_wmma_f32_16x16x32_bf16(false, a0, false, b0, (short)0, acc00, false, false);
        acc01 = __builtin_amdgcn_wmma_f32_16x16x32_bf16(false, a0, false, b1, (short)0, acc01, false, false);
        acc10 = __builtin_amdgcn_wmma_f32_16x16x32_bf16(false, a1, false, b0, (short)0, acc10, false, false);
        acc11 = __builtin_amdgcn_wmma_f32_16x16x32_bf16(false, a1, false, b1, (short)0, acc11, false, false);

        __syncthreads();                             // protect buffer reuse
    }

    const long long m0 = mblk + wm * 32;
    const long long n0 = nblk + wn * 32;
    const long long cols[2] = { n0 + l16, n0 + 16 + l16 };
    f32x8 accr[2][2] = { { acc00, acc01 }, { acc10, acc11 } };
#pragma unroll
    for (int i = 0; i < 2; ++i) {
#pragma unroll
        for (int j = 0; j < 2; ++j) {
            long long col = cols[j];
            if (col < N) {
                long long rbase = m0 + 16 * i + 8 * half;
#pragma unroll
                for (int v = 0; v < 8; ++v) {
                    long long idx = (rbase + v) * (long long)N + col;
                    if (OUT_BF16) ((bf16*)Cout)[idx]  = (bf16)accr[i][j][v];
                    else          ((float*)Cout)[idx] = accr[i][j][v];
                }
            }
        }
    }
}

// ---------------------------------------------------------------------------
// Depthwise causal conv (k=4) + bias + SiLU. x_inner = first 2048 cols of xz.
// ---------------------------------------------------------------------------
__global__ __launch_bounds__(256) void conv_silu_kernel(const bf16* __restrict__ xz,
                                                        const float* __restrict__ cw,
                                                        const float* __restrict__ cb,
                                                        bf16* __restrict__ xconv) {
    long long idx = (long long)blockIdx.x * 256 + threadIdx.x;
    if (idx >= (long long)kTOKENS * kDINNER) return;
    int d          = (int)(idx % kDINNER);
    long long tok  = idx / kDINNER;
    int t          = (int)(tok % kSEQ);
    long long b    = tok / kSEQ;

    float acc = cb[d];
#pragma unroll
    for (int k = 0; k < kDCONV; ++k) {
        int tt = t - (kDCONV - 1) + k;
        if (tt >= 0)
            acc += cw[d * kDCONV + k] *
                   (float)xz[((long long)b * kSEQ + tt) * (2 * kDINNER) + d];
    }
    float s = acc / (1.f + __expf(-acc));   // SiLU
    xconv[tok * kDINNER + d] = (bf16)s;
}

// ---------------------------------------------------------------------------
// Selective scan. One independent wave32 per (batch, channel); no barriers.
// Lane owns states s = lane + 32k  -> stride-32 coalesced B/C loads.
// dt = softplus(dt_raw * w_d + b_d) fused (num_heads == 1).
// ---------------------------------------------------------------------------
__global__ __launch_bounds__(256) void scan_kernel(const float* __restrict__ xdbl,  // [T,257]
                                                   const bf16*  __restrict__ xconv, // [T,2048]
                                                   const float* __restrict__ dtw,   // [2048]
                                                   const float* __restrict__ dtb,   // [2048]
                                                   const float* __restrict__ Alog,  // [128]
                                                   float* __restrict__ y) {         // [T,2048]
    const int lane = threadIdx.x & 31;
    const int wave = threadIdx.x >> 5;                 // 0..7
    const int b    = blockIdx.x / (kDINNER / 8);
    const int dg   = blockIdx.x % (kDINNER / 8);
    const int d    = dg * 8 + wave;

    const float w_d = dtw[d];
    const float b_d = dtb[d];

    float Am[4], h[4] = {0.f, 0.f, 0.f, 0.f};
#pragma unroll
    for (int k = 0; k < 4; ++k) Am[k] = -__expf(Alog[lane + 32 * k]);

    const float* rowb = xdbl + (long long)b * kSEQ * kNDBL;
    for (int t = 0; t < kSEQ; ++t) {
        const float* row = rowb + (long long)t * kNDBL;
        float dtr = row[0];
        float Bv[4], Cv[4];
#pragma unroll
        for (int k = 0; k < 4; ++k) {
            Bv[k] = row[1 + lane + 32 * k];
            Cv[k] = row[1 + kDSTATE + lane + 32 * k];
        }
        long long tok = (long long)b * kSEQ + t;
        float pre = dtr * w_d + b_d;
        float dt  = (pre > 20.f) ? pre : log1pf(__expf(pre));   // softplus
        float u   = (float)xconv[tok * kDINNER + d];
        float dtu = dt * u;

        float accv = 0.f;
#pragma unroll
        for (int k = 0; k < 4; ++k) {
            float dA = __expf(dt * Am[k]);
            h[k]     = dA * h[k] + dtu * Bv[k];
            accv    += h[k] * Cv[k];
        }
#pragma unroll
        for (int off = 16; off > 0; off >>= 1) accv += __shfl_xor(accv, off, 32);
        if (lane == 0) y[tok * kDINNER + d] = accv;
    }
}

// ---------------------------------------------------------------------------
// RMS norm over d_inner, * norm_w, * silu(z). One block per token.
// ---------------------------------------------------------------------------
__global__ __launch_bounds__(256) void norm_gate_kernel(const float* __restrict__ y,
                                                        const bf16*  __restrict__ xz,
                                                        const float* __restrict__ normw,
                                                        bf16* __restrict__ yn) {
    const long long tok = blockIdx.x;
    const float* yr = y  + tok * kDINNER;
    const bf16*  zr = xz + tok * (2 * kDINNER) + kDINNER;
    __shared__ float red[8];

    float ss = 0.f;
    for (int i = threadIdx.x; i < kDINNER; i += 256) { float v = yr[i]; ss += v * v; }
#pragma unroll
    for (int off = 16; off > 0; off >>= 1) ss += __shfl_xor(ss, off, 32);
    if ((threadIdx.x & 31) == 0) red[threadIdx.x >> 5] = ss;
    __syncthreads();
    float tot = 0.f;
#pragma unroll
    for (int wv = 0; wv < 8; ++wv) tot += red[wv];

    float r = rsqrtf(tot / (float)kDINNER + 1.1920929e-7f);
    for (int i = threadIdx.x; i < kDINNER; i += 256) {
        float v  = yr[i] * r * normw[i];
        float z  = (float)zr[i];
        float sz = z / (1.f + __expf(-z));
        yn[tok * kDINNER + i] = (bf16)(v * sz);
    }
}

// ---------------------------------------------------------------------------
// host launcher
// ---------------------------------------------------------------------------
extern "C" void kernel_launch(void* const* d_in, const int* in_sizes, int n_in,
                              void* d_out, int out_size, void* d_ws, size_t ws_size,
                              hipStream_t stream) {
    (void)in_sizes; (void)n_in; (void)out_size; (void)ws_size;

    const float* x       = (const float*)d_in[0];  // [4,4096,1024]
    const float* in_w    = (const float*)d_in[1];  // [4096,1024]
    const float* conv_w  = (const float*)d_in[2];  // [2048,1,4]
    const float* conv_b  = (const float*)d_in[3];  // [2048]
    const float* xproj_w = (const float*)d_in[4];  // [257,2048]
    const float* dt_w    = (const float*)d_in[5];  // [2048,1]
    const float* dt_b    = (const float*)d_in[6];  // [2048]
    const float* A_log   = (const float*)d_in[7];  // [1,128]
    const float* norm_w  = (const float*)d_in[8];  // [2048]
    const float* out_w   = (const float*)d_in[9];  // [1024,2048]
    float* out = (float*)d_out;                    // [4,4096,1024]

    // workspace layout (256-byte aligned regions)
    size_t off = 0;
    auto carve = [&](size_t bytes) { size_t o = off; off += (bytes + 255) & ~(size_t)255; return o; };
    char* ws = (char*)d_ws;
    bf16*  xbf    = (bf16*)(ws + carve((size_t)kTOKENS * kDIM * 2));           // 32 MB
    bf16*  winbf  = (bf16*)(ws + carve((size_t)2 * kDINNER * kDIM * 2));       // 8 MB
    bf16*  wxbf   = (bf16*)(ws + carve((size_t)kNDBL * kDINNER * 2));          // 1 MB
    bf16*  woutbf = (bf16*)(ws + carve((size_t)kDIM * kDINNER * 2));           // 4 MB
    bf16*  xzbf   = (bf16*)(ws + carve((size_t)kTOKENS * 2 * kDINNER * 2));    // 128 MB
    bf16*  xcbf   = (bf16*)(ws + carve((size_t)kTOKENS * kDINNER * 2));        // 64 MB
    float* xdbl   = (float*)(ws + carve((size_t)kTOKENS * kNDBL * 4));         // 16 MB
    float* yf     = (float*)(ws + carve((size_t)kTOKENS * kDINNER * 4));       // 128 MB
    bf16*  ynbf   = (bf16*)(ws + carve((size_t)kTOKENS * kDINNER * 2));        // 64 MB

    auto cvt = [&](const float* src, bf16* dst, long long n) {
        f32_to_bf16_kernel<<<dim3((unsigned)((n + 255) / 256)), dim3(256), 0, stream>>>(src, dst, n);
    };
    cvt(x,       xbf,    (long long)kTOKENS * kDIM);
    cvt(in_w,    winbf,  (long long)2 * kDINNER * kDIM);
    cvt(xproj_w, wxbf,   (long long)kNDBL * kDINNER);
    cvt(out_w,   woutbf, (long long)kDIM * kDINNER);

    // GEMM1: xz[T, 4096] (bf16) = x[T,1024] * in_proj_w^T
    {
        dim3 g((2 * kDINNER) / 64, kTOKENS / 128);
        wmma_gemm_kernel<true><<<g, dim3(256), 0, stream>>>(xbf, winbf, xzbf,
                                                            kTOKENS, 2 * kDINNER, kDIM);
    }
    // depthwise conv + SiLU -> xconv bf16
    {
        long long n = (long long)kTOKENS * kDINNER;
        conv_silu_kernel<<<dim3((unsigned)((n + 255) / 256)), dim3(256), 0, stream>>>(
            xzbf, conv_w, conv_b, xcbf);
    }
    // GEMM2: x_dbl[T, 257] (f32) = xconv[T,2048] * x_proj_w^T
    {
        dim3 g((kNDBL + 63) / 64, kTOKENS / 128);
        wmma_gemm_kernel<false><<<g, dim3(256), 0, stream>>>(xcbf, wxbf, xdbl,
                                                             kTOKENS, kNDBL, kDINNER);
    }
    // selective scan -> y f32
    scan_kernel<<<dim3(kBATCH * (kDINNER / 8)), dim3(256), 0, stream>>>(
        xdbl, xcbf, dt_w, dt_b, A_log, yf);
    // RMS norm * norm_w * silu(z) -> ynorm bf16
    norm_gate_kernel<<<dim3(kTOKENS), dim3(256), 0, stream>>>(yf, xzbf, norm_w, ynbf);
    // GEMM3: out[T, 1024] (f32) = ynorm[T,2048] * out_proj_w^T
    {
        dim3 g(kDIM / 64, kTOKENS / 128);
        wmma_gemm_kernel<false><<<g, dim3(256), 0, stream>>>(ynbf, woutbf, out,
                                                             kTOKENS, kDIM, kDINNER);
    }
}
